// Model_76364518523249
// MI455X (gfx1250) — matrix operand order repaired and checked by reading, not compiled, not run on hardware
//
#include <hip/hip_runtime.h>

// ---------------------------------------------------------------------------
// Problem dimensions
// ---------------------------------------------------------------------------
#define DI     32
#define DM     256
#define DB     512
#define CTX    96
#define BQ     1024
#define NTRAIN 100000
#define TCH    64                  // candidates per top-k chunk
#define NPAD   100096              // 1564 chunks * 64 candidates
#define NCHUNK (NPAD / TCH)

typedef __bf16 bf16;
typedef __attribute__((ext_vector_type(16))) __bf16 bf16x16;
typedef __attribute__((ext_vector_type(8)))  float  f32x8;

#define F32X8_ZERO {0.f, 0.f, 0.f, 0.f, 0.f, 0.f, 0.f, 0.f}

// ---------------------------------------------------------------------------
// WMMA helpers (gfx1250, wave32, 16x16x32 bf16 -> f32)
// ---------------------------------------------------------------------------
__device__ inline f32x8 wmma_bf16(bf16x16 a, bf16x16 b, f32x8 c) {
  return __builtin_amdgcn_wmma_f32_16x16x32_bf16(
      /*neg_a=*/false, a, /*neg_b=*/false, b,
      /*c_mod=*/(short)0, c, /*reuse_a=*/false, /*reuse_b=*/false);
}

// A fragment (16x32 bf16, M x K) from row-major bf16 [16][lda] in LDS.
// lane(0-15)=row M (lanes 16-31 same M); v0..3 -> K=2v+8*half, v4..7 -> +16.
__device__ inline bf16x16 ldA_bf(const bf16* A, int lda, int kb) {
  const int lane = threadIdx.x & 31;
  const int hh = lane >> 4, m = lane & 15;
  const bf16* p = A + m * lda + kb + 8 * hh;
  bf16x16 a;
#pragma unroll
  for (int v = 0; v < 4; ++v) {
    a[2 * v]         = p[2 * v];
    a[2 * v + 1]     = p[2 * v + 1];
    a[8 + 2 * v]     = p[16 + 2 * v];
    a[8 + 2 * v + 1] = p[16 + 2 * v + 1];
  }
  return a;
}

// Same but source is f32 LDS (convert on load).
__device__ inline bf16x16 ldA_f32(const float* A, int lda, int kb) {
  const int lane = threadIdx.x & 31;
  const int hh = lane >> 4, m = lane & 15;
  const float* p = A + m * lda + kb + 8 * hh;
  bf16x16 a;
#pragma unroll
  for (int v = 0; v < 4; ++v) {
    a[2 * v]         = (bf16)p[2 * v];
    a[2 * v + 1]     = (bf16)p[2 * v + 1];
    a[8 + 2 * v]     = (bf16)p[16 + 2 * v];
    a[8 + 2 * v + 1] = (bf16)p[16 + 2 * v + 1];
  }
  return a;
}

// B fragment (32x16 bf16, K x N) with B[k][n] = W[n][k], W row-major [N][K].
// lane(0-15)=col N; v0..7 -> K=v+8*half, v8..15 -> K=16+8*half+(v-8).
__device__ inline bf16x16 ldB(const bf16* W, int K, int n0, int kb) {
  const int lane = threadIdx.x & 31;
  const int hh = lane >> 4, n = n0 + (lane & 15);
  const bf16* p = W + (size_t)n * K + kb + 8 * hh;
  bf16x16 b;
#pragma unroll
  for (int v = 0; v < 8; ++v) { b[v] = p[v]; b[8 + v] = p[16 + v]; }
  return b;
}

__device__ inline void ln_stats(const float* sx, float* smu, float* srs, int tid) {
  if (tid < 16) {
    float s = 0.f;
    for (int d = 0; d < DM; ++d) s += sx[tid * DM + d];
    float mu = s * (1.f / DM);
    float v = 0.f;
    for (int d = 0; d < DM; ++d) { float t = sx[tid * DM + d] - mu; v += t * t; }
    smu[tid] = mu;
    srs[tid] = rsqrtf(v * (1.f / DM) + 1e-5f);
  }
}

// Async global->LDS staging (ASYNCcnt-tracked, CDNA5).  Each calling thread
// copies `bytes`/16 x b128 from contiguous global memory to contiguous LDS.
__device__ inline void async_copy_b128(unsigned lds_addr, unsigned long long gaddr, int n128) {
  for (int i = 0; i < n128; ++i) {
    asm volatile("global_load_async_to_lds_b128 %0, %1, off"
                 :: "v"(lds_addr + (unsigned)(i * 16)),
                    "v"(gaddr + (unsigned long long)(i * 16))
                 : "memory");
  }
}
__device__ inline void wait_async0() {
  asm volatile("s_wait_asynccnt 0" ::: "memory");
}

// ---------------------------------------------------------------------------
// Prep kernels
// ---------------------------------------------------------------------------
__global__ void cvt_bf16(const float* __restrict__ src, bf16* __restrict__ dst, int n) {
  int i = blockIdx.x * blockDim.x + threadIdx.x;
  if (i < n) dst[i] = (bf16)src[i];
}

__global__ void pad_kernel(bf16* __restrict__ ckbf, float* __restrict__ sqc) {
  int i = blockIdx.x * blockDim.x + threadIdx.x;
  const int prows = NPAD - NTRAIN;
  if (i < prows * DM) ckbf[(size_t)NTRAIN * DM + i] = (bf16)0.f;
  if (i < prows) sqc[NTRAIN + i] = 1.0e30f;
}

// ---------------------------------------------------------------------------
// Encoder: x = Lin(X) + resblock; k = LN(x) @ Wk^T + bk
// 128 threads (4 waves) per 16-row tile.
// ---------------------------------------------------------------------------
__global__ __launch_bounds__(128)
void encode_kernel(const float* __restrict__ Xin,
                   const bf16* __restrict__ Win, const float* __restrict__ b_in,
                   const bf16* __restrict__ W0a, const float* __restrict__ b0a,
                   const bf16* __restrict__ W0b, const float* __restrict__ b0b,
                   const float* __restrict__ g_mix, const float* __restrict__ be_mix,
                   const bf16* __restrict__ Wk, const float* __restrict__ b_k,
                   float* __restrict__ x_out, float* __restrict__ k_out,
                   bf16* __restrict__ kbf_out, float* __restrict__ sq_out) {
  __shared__ bf16  sA[16 * DI];
  __shared__ float sx[16 * DM];
  __shared__ bf16  sT[16 * DB];   // also reused as bf16 LN buffer [16*DM]
  __shared__ float smu[16], srs[16], ssq[16];

  const int tid = threadIdx.x, lane = tid & 31, wave = tid >> 5;
  const int hh = lane >> 4;
  const int row0 = blockIdx.x * 16;

  for (int t = tid; t < 16 * DI; t += 128) {
    int r = t / DI, c = t % DI;
    sA[t] = (bf16)Xin[(size_t)(row0 + r) * DI + c];
  }
  __syncthreads();

  // GEMM1: x = X @ Win^T + b_in   (K = 32, one A fragment reused for 4 tiles)
  {
    bf16x16 a0 = ldA_bf(sA, DI, 0);
#pragma unroll
    for (int j = 0; j < 4; ++j) {
      int n0 = (wave * 4 + j) * 16;
      f32x8 acc = F32X8_ZERO;
      acc = wmma_bf16(a0, ldB(Win, DI, n0, 0), acc);
      int n = n0 + (lane & 15);
      float bias = b_in[n];
#pragma unroll
      for (int i = 0; i < 8; ++i) sx[(i + 8 * hh) * DM + n] = acc[i] + bias;
    }
  }
  __syncthreads();

  // GEMM2: t = relu(x @ W0a^T + b0a)   (A fragments hoisted, reused 8x)
  {
    bf16x16 af[8];
#pragma unroll
    for (int kb = 0; kb < 8; ++kb) af[kb] = ldA_f32(sx, DM, kb * 32);
#pragma unroll
    for (int j = 0; j < 8; ++j) {
      int n0 = (wave * 8 + j) * 16;
      f32x8 acc = F32X8_ZERO;
#pragma unroll
      for (int kb = 0; kb < 8; ++kb)
        acc = wmma_bf16(af[kb], ldB(W0a, DM, n0, kb * 32), acc);
      int n = n0 + (lane & 15);
      float bias = b0a[n];
#pragma unroll
      for (int i = 0; i < 8; ++i) {
        float v = acc[i] + bias;
        sT[(i + 8 * hh) * DB + n] = (bf16)(v > 0.f ? v : 0.f);
      }
    }
  }
  __syncthreads();

  // GEMM3: x += t @ W0b^T + b0b   (K=512 as two halves, A reused 4x per half)
  {
    f32x8 acc[4];
#pragma unroll
    for (int j = 0; j < 4; ++j) acc[j] = (f32x8)F32X8_ZERO;
#pragma unroll
    for (int h = 0; h < 2; ++h) {
      bf16x16 af[8];
#pragma unroll
      for (int kb = 0; kb < 8; ++kb) af[kb] = ldA_bf(sT, DB, h * 256 + kb * 32);
#pragma unroll
      for (int j = 0; j < 4; ++j) {
        int n0 = (wave * 4 + j) * 16;
#pragma unroll
        for (int kb = 0; kb < 8; ++kb)
          acc[j] = wmma_bf16(af[kb], ldB(W0b, DB, n0, h * 256 + kb * 32), acc[j]);
      }
    }
#pragma unroll
    for (int j = 0; j < 4; ++j) {
      int n0 = (wave * 4 + j) * 16;
      int n = n0 + (lane & 15);
      float bias = b0b[n];
#pragma unroll
      for (int i = 0; i < 8; ++i) sx[(i + 8 * hh) * DM + n] += acc[j][i] + bias;
    }
  }
  __syncthreads();

  // LayerNorm(g_mix, be_mix) -> bf16 (reuse sT storage)
  ln_stats(sx, smu, srs, tid);
  if (tid < 16) ssq[tid] = 0.f;
  __syncthreads();
  bf16* sLN = sT;
  for (int t = tid; t < 16 * DM; t += 128) {
    int r = t / DM, d = t % DM;
    sLN[t] = (bf16)((sx[t] - smu[r]) * srs[r] * g_mix[d] + be_mix[d]);
  }
  __syncthreads();

  // GEMM4: k = LN @ Wk^T + b_k   (A hoisted, reused 4x)
  {
    bf16x16 af[8];
#pragma unroll
    for (int kb = 0; kb < 8; ++kb) af[kb] = ldA_bf(sLN, DM, kb * 32);
#pragma unroll
    for (int j = 0; j < 4; ++j) {
      int n0 = (wave * 4 + j) * 16;
      f32x8 acc = F32X8_ZERO;
#pragma unroll
      for (int kb = 0; kb < 8; ++kb)
        acc = wmma_bf16(af[kb], ldB(Wk, DM, n0, kb * 32), acc);
      int n = n0 + (lane & 15);
      float bias = b_k[n];
#pragma unroll
      for (int i = 0; i < 8; ++i) {
        int m = i + 8 * hh;
        float kv = acc[i] + bias;
        size_t g = (size_t)(row0 + m) * DM + n;
        k_out[g] = kv;
        kbf_out[g] = (bf16)kv;
        atomicAdd(&ssq[m], kv * kv);
      }
    }
  }
  __syncthreads();
  if (tid < 16) sq_out[row0 + tid] = ssq[tid];
  if (x_out) {
    for (int t = tid; t < 16 * DM; t += 128)
      x_out[(size_t)row0 * DM + t] = sx[t];
  }
}

// ---------------------------------------------------------------------------
// Top-k: 128 threads (4 waves) per 16-query tile; 64-candidate chunks staged
// into LDS with async DMA, selection overlapped with next chunk's copy.
// ---------------------------------------------------------------------------
__device__ inline void rescan_min(const float* sts, int q, int lane, float& cm, int& mp) {
  float mv = sts[q * CTX + lane];
  int mpi = lane;
  float v2 = sts[q * CTX + lane + 32];
  if (v2 < mv) { mv = v2; mpi = lane + 32; }
  float v3 = sts[q * CTX + lane + 64];
  if (v3 < mv) { mv = v3; mpi = lane + 64; }
  for (int off = 16; off > 0; off >>= 1) {
    float ov = __shfl_down(mv, off);
    int op = __shfl_down(mpi, off);
    if (ov < mv) { mv = ov; mpi = op; }
  }
  cm = __shfl(mv, 0);
  mp = __shfl(mpi, 0);
}

__global__ __launch_bounds__(128)
void topk_kernel(const bf16* __restrict__ kbf, const float* __restrict__ sq_k,
                 const bf16* __restrict__ ckbf, const float* __restrict__ sq_c,
                 int* __restrict__ idx_out, float* __restrict__ sims_out) {
  __shared__ bf16  skq[16 * DM];        // 8 KB
  __shared__ bf16  scand[TCH * DM];     // 32 KB candidate chunk
  __shared__ float ssc[16 * TCH];       // 4 KB scores
  __shared__ float sts[16 * CTX];       // 6 KB top values
  __shared__ int   sti[16 * CTX];       // 6 KB top indices
  __shared__ float ssqk[16];

  const int tid = threadIdx.x, lane = tid & 31, wave = tid >> 5;
  const int hh = lane >> 4;
  const int q0 = blockIdx.x * 16;

  // async-stage the query tile (8 KB; 64 B per thread)
  async_copy_b128((unsigned)(uintptr_t)&skq[0] + (unsigned)tid * 64u,
                  (unsigned long long)(uintptr_t)(kbf + (size_t)q0 * DM) +
                      (unsigned long long)tid * 64ull, 4);
  // async-stage candidate chunk 0 (32 KB; 256 B per thread)
  async_copy_b128((unsigned)(uintptr_t)&scand[0] + (unsigned)tid * 256u,
                  (unsigned long long)(uintptr_t)ckbf + (unsigned long long)tid * 256ull, 16);

  if (tid < 16) ssqk[tid] = sq_k[q0 + tid];
  for (int t = tid; t < 16 * CTX; t += 128) { sts[t] = -3.0e38f; sti[t] = 0; }
  wait_async0();
  __syncthreads();

  bf16x16 afr[8];
#pragma unroll
  for (int kb = 0; kb < 8; ++kb) afr[kb] = ldA_bf(skq, DM, kb * 32);

  float curmin[4];
  int minpos[4];
#pragma unroll
  for (int qq = 0; qq < 4; ++qq) { curmin[qq] = -3.0e38f; minpos[qq] = 0; }

  for (int chunk = 0; chunk < NCHUNK; ++chunk) {
    // score this chunk: wave w covers local candidates [16w, 16w+16)
    const int c0 = wave * 16;
    f32x8 acc = F32X8_ZERO;
#pragma unroll
    for (int kb = 0; kb < 8; ++kb)
      acc = wmma_bf16(afr[kb], ldB(scand, DM, c0, kb * 32), acc);

    const int nl = lane & 15;
    const float sqc = sq_c[chunk * TCH + c0 + nl];
#pragma unroll
    for (int i = 0; i < 8; ++i) {
      int m = i + 8 * hh;
      ssc[m * TCH + c0 + nl] = 2.f * acc[i] - sqc - ssqk[m];
    }
    __syncthreads();   // chunk reads + score writes complete

    // overlap: start DMA of the next chunk into scand while we do selection
    if (chunk + 1 < NCHUNK)
      async_copy_b128((unsigned)(uintptr_t)&scand[0] + (unsigned)tid * 256u,
                      (unsigned long long)(uintptr_t)
                          (ckbf + (size_t)(chunk + 1) * TCH * DM) +
                          (unsigned long long)tid * 256ull, 16);

    // selection: wave owns queries 4*wave .. 4*wave+3
#pragma unroll
    for (int qq = 0; qq < 4; ++qq) {
      const int q = 4 * wave + qq;
      float cm = curmin[qq];
      int mp = minpos[qq];
#pragma unroll
      for (int j = 0; j < 2; ++j) {
        float s = ssc[q * TCH + lane + 32 * j];
        unsigned long long mask = __ballot(s > cm);
        while (mask) {
          int l = __ffsll(mask) - 1;
          mask &= mask - 1;
          float sl = __shfl(s, l);
          if (sl > cm) {
            if (lane == 0) {
              sts[q * CTX + mp] = sl;
              sti[q * CTX + mp] = chunk * TCH + 32 * j + l;
            }
            __builtin_amdgcn_wave_barrier();
            rescan_min(sts, q, lane, cm, mp);
          }
        }
      }
      curmin[qq] = cm;
      minpos[qq] = mp;
    }

    if (chunk + 1 < NCHUNK) wait_async0();
    __syncthreads();
  }

#pragma unroll
  for (int qq = 0; qq < 4; ++qq) {
    int q = 4 * wave + qq;
    for (int c = lane; c < CTX; c += 32) {
      sims_out[(size_t)(q0 + q) * CTX + c] = sts[q * CTX + c];
      idx_out[(size_t)(q0 + q) * CTX + c] = sti[q * CTX + c];
    }
  }
}

// ---------------------------------------------------------------------------
// Aggregate: per-query softmax + pair-MLP (256->512->256) + weighted sum.
// 256 threads (8 waves) per query; 6 M-tiles of 16 pairs.
// ---------------------------------------------------------------------------
__global__ __launch_bounds__(256)
void aggregate_kernel(const float* __restrict__ k_f32,
                      const float* __restrict__ ck_f32,
                      const float* __restrict__ y_train,
                      const int* __restrict__ idx_in,
                      const float* __restrict__ sims_in,
                      const bf16* __restrict__ Wt1, const float* __restrict__ b_t1,
                      const bf16* __restrict__ Wt2,
                      const float* __restrict__ w_lab, const float* __restrict__ b_lab,
                      const float* __restrict__ x_enc,
                      float* __restrict__ x2_out) {
  __shared__ float sprob[CTX];
  __shared__ int   sidx[CTX];
  __shared__ float skq[DM];
  __shared__ bf16  sdiff[16 * DM];
  __shared__ bf16  st1[16 * DB];
  __shared__ float sctx[DM];
  __shared__ float s_py;

  const int tid = threadIdx.x, lane = tid & 31, wave = tid >> 5;
  const int hh = lane >> 4;
  const int b = blockIdx.x;

  if (tid < CTX) {
    sprob[tid] = sims_in[(size_t)b * CTX + tid];
    int ix = idx_in[(size_t)b * CTX + tid];
    sidx[tid] = (ix >= 0 && ix < NTRAIN) ? ix : 0;
  }
  if (tid < DM) {
    skq[tid] = k_f32[(size_t)b * DM + tid];
    sctx[tid] = 0.f;
  }
  __syncthreads();

  if (tid == 0) {
    float mx = -3.0e38f;
    for (int c = 0; c < CTX; ++c) mx = fmaxf(mx, sprob[c]);
    float sum = 0.f;
    for (int c = 0; c < CTX; ++c) { float e = __expf(sprob[c] - mx); sprob[c] = e; sum += e; }
    float inv = 1.f / sum, py = 0.f;
    for (int c = 0; c < CTX; ++c) {
      sprob[c] *= inv;
      py += sprob[c] * y_train[sidx[c]];
    }
    s_py = py;
  }
  __syncthreads();

  for (int ct = 0; ct < CTX / 16; ++ct) {
    for (int t = tid; t < 16 * DM; t += 256) {
      int r = t >> 8, d = t & (DM - 1);
      const float* ckrow = ck_f32 + (size_t)sidx[ct * 16 + r] * DM;
      sdiff[t] = (bf16)(skq[d] - ckrow[d]);
    }
    __syncthreads();

    // t1 = relu(diff @ Wt1^T + b_t1)   (A hoisted, reused 4x)
    {
      bf16x16 af[8];
#pragma unroll
      for (int kb = 0; kb < 8; ++kb) af[kb] = ldA_bf(sdiff, DM, kb * 32);
#pragma unroll
      for (int j = 0; j < 4; ++j) {
        int n0 = (wave + 8 * j) * 16;
        f32x8 acc = F32X8_ZERO;
#pragma unroll
        for (int kb = 0; kb < 8; ++kb)
          acc = wmma_bf16(af[kb], ldB(Wt1, DM, n0, kb * 32), acc);
        int n = n0 + (lane & 15);
        float bias = b_t1[n];
#pragma unroll
        for (int i = 0; i < 8; ++i) {
          float v = acc[i] + bias;
          st1[(i + 8 * hh) * DB + n] = (bf16)(v > 0.f ? v : 0.f);
        }
      }
    }
    __syncthreads();

    // v = t1 @ Wt2^T ; probability-weighted accumulate into sctx
    {
      f32x8 acc[2];
#pragma unroll
      for (int j = 0; j < 2; ++j) acc[j] = (f32x8)F32X8_ZERO;
#pragma unroll
      for (int h = 0; h < 2; ++h) {
        bf16x16 af[8];
#pragma unroll
        for (int kb = 0; kb < 8; ++kb) af[kb] = ldA_bf(st1, DB, h * 256 + kb * 32);
#pragma unroll
        for (int j = 0; j < 2; ++j) {
          int n0 = (wave + 8 * j) * 16;
#pragma unroll
          for (int kb = 0; kb < 8; ++kb)
            acc[j] = wmma_bf16(af[kb], ldB(Wt2, DB, n0, h * 256 + kb * 32), acc[j]);
        }
      }
#pragma unroll
      for (int j = 0; j < 2; ++j) {
        int n0 = (wave + 8 * j) * 16;
        int n = n0 + (lane & 15);
        float part = 0.f;
#pragma unroll
        for (int i = 0; i < 8; ++i) part += sprob[ct * 16 + i + 8 * hh] * acc[j][i];
        atomicAdd(&sctx[n], part);
      }
    }
    __syncthreads();
  }

  if (tid < DM) {
    float v = x_enc[(size_t)b * DM + tid] + sctx[tid] + s_py * w_lab[tid] + b_lab[tid];
    x2_out[(size_t)b * DM + tid] = v;
  }
}

// ---------------------------------------------------------------------------
// Predictor block (prenorm) + head. 128 threads / 16 rows.
// ---------------------------------------------------------------------------
__global__ __launch_bounds__(128)
void predictor_kernel(const float* __restrict__ x2,
                      const float* __restrict__ g_p, const float* __restrict__ be_p,
                      const bf16* __restrict__ W1a, const float* __restrict__ b1a,
                      const bf16* __restrict__ W1b, const float* __restrict__ b1b,
                      const float* __restrict__ g_h, const float* __restrict__ be_h,
                      const float* __restrict__ W_h, const float* __restrict__ b_h,
                      float* __restrict__ out) {
  __shared__ float sx[16 * DM];
  __shared__ bf16  sLN[16 * DM];
  __shared__ bf16  sT[16 * DB];
  __shared__ float smu[16], srs[16];
  __shared__ float shp[16 * 8];

  const int tid = threadIdx.x, lane = tid & 31, wave = tid >> 5;
  const int hh = lane >> 4;
  const int row0 = blockIdx.x * 16;

  for (int t = tid; t < 16 * DM; t += 128) sx[t] = x2[(size_t)row0 * DM + t];
  __syncthreads();

  ln_stats(sx, smu, srs, tid);
  __syncthreads();
  for (int t = tid; t < 16 * DM; t += 128) {
    int r = t >> 8, d = t & (DM - 1);
    sLN[t] = (bf16)((sx[t] - smu[r]) * srs[r] * g_p[d] + be_p[d]);
  }
  __syncthreads();

  // t = relu(LN @ W1a^T + b1a)   (A hoisted, reused 8x)
  {
    bf16x16 af[8];
#pragma unroll
    for (int kb = 0; kb < 8; ++kb) af[kb] = ldA_bf(sLN, DM, kb * 32);
#pragma unroll
    for (int j = 0; j < 8; ++j) {
      int n0 = (wave * 8 + j) * 16;
      f32x8 acc = F32X8_ZERO;
#pragma unroll
      for (int kb = 0; kb < 8; ++kb)
        acc = wmma_bf16(af[kb], ldB(W1a, DM, n0, kb * 32), acc);
      int n = n0 + (lane & 15);
      float bias = b1a[n];
#pragma unroll
      for (int i = 0; i < 8; ++i) {
        float v = acc[i] + bias;
        sT[(i + 8 * hh) * DB + n] = (bf16)(v > 0.f ? v : 0.f);
      }
    }
  }
  __syncthreads();

  // x3 = x2 + t @ W1b^T + b1b   (K=512 as two halves)
  {
    f32x8 acc[4];
#pragma unroll
    for (int j = 0; j < 4; ++j) acc[j] = (f32x8)F32X8_ZERO;
#pragma unroll
    for (int h = 0; h < 2; ++h) {
      bf16x16 af[8];
#pragma unroll
      for (int kb = 0; kb < 8; ++kb) af[kb] = ldA_bf(sT, DB, h * 256 + kb * 32);
#pragma unroll
      for (int j = 0; j < 4; ++j) {
        int n0 = (wave * 4 + j) * 16;
#pragma unroll
        for (int kb = 0; kb < 8; ++kb)
          acc[j] = wmma_bf16(af[kb], ldB(W1b, DB, n0, h * 256 + kb * 32), acc[j]);
      }
    }
#pragma unroll
    for (int j = 0; j < 4; ++j) {
      int n0 = (wave * 4 + j) * 16;
      int n = n0 + (lane & 15);
      float bias = b1b[n];
#pragma unroll
      for (int i = 0; i < 8; ++i) sx[(i + 8 * hh) * DM + n] += acc[j][i] + bias;
    }
  }
  __syncthreads();

  ln_stats(sx, smu, srs, tid);
  __syncthreads();

  {
    int r = tid >> 3, p = tid & 7;
    float s = 0.f;
    for (int d = p * 32; d < p * 32 + 32; ++d) {
      float v = (sx[r * DM + d] - smu[r]) * srs[r] * g_h[d] + be_h[d];
      v = v > 0.f ? v : 0.f;
      s += v * W_h[d];
    }
    shp[tid] = s;
  }
  __syncthreads();
  if (tid < 16) {
    float s = 0.f;
    for (int p = 0; p < 8; ++p) s += shp[tid * 8 + p];
    out[row0 + tid] = s + b_h[0];
  }
}

// ---------------------------------------------------------------------------
// Host launcher
// ---------------------------------------------------------------------------
extern "C" void kernel_launch(void* const* d_in, const int* in_sizes, int n_in,
                              void* d_out, int out_size, void* d_ws, size_t ws_size,
                              hipStream_t stream) {
  (void)in_sizes; (void)n_in; (void)out_size; (void)ws_size;
  const float* X      = (const float*)d_in[0];
  const float* Xtr    = (const float*)d_in[1];
  const float* ytr    = (const float*)d_in[2];
  const float* W_in   = (const float*)d_in[3];
  const float* b_in   = (const float*)d_in[4];
  const float* W0a    = (const float*)d_in[5];
  const float* b0a    = (const float*)d_in[6];
  const float* W0b    = (const float*)d_in[7];
  const float* b0b    = (const float*)d_in[8];
  const float* g_mix  = (const float*)d_in[9];
  const float* be_mix = (const float*)d_in[10];
  const float* W_k    = (const float*)d_in[11];
  const float* b_k    = (const float*)d_in[12];
  const float* w_lab  = (const float*)d_in[13];
  const float* b_lab  = (const float*)d_in[14];
  const float* W_t1   = (const float*)d_in[15];
  const float* b_t1   = (const float*)d_in[16];
  const float* W_t2   = (const float*)d_in[17];
  const float* g_p    = (const float*)d_in[18];
  const float* be_p   = (const float*)d_in[19];
  const float* W1a    = (const float*)d_in[20];
  const float* b1a    = (const float*)d_in[21];
  const float* W1b    = (const float*)d_in[22];
  const float* b1b    = (const float*)d_in[23];
  const float* g_h    = (const float*)d_in[24];
  const float* be_h   = (const float*)d_in[25];
  const float* W_h    = (const float*)d_in[26];
  const float* b_h    = (const float*)d_in[27];
  float* out = (float*)d_out;

  char* ws = (char*)d_ws;
  size_t o = 0;
  auto take = [&](size_t bytes) -> char* {
    char* p = ws + o;
    o = (o + bytes + 255) & ~(size_t)255;
    return p;
  };

  bf16* Win_bf = (bf16*)take((size_t)DM * DI * 2);
  bf16* W0a_bf = (bf16*)take((size_t)DB * DM * 2);
  bf16* W0b_bf = (bf16*)take((size_t)DM * DB * 2);
  bf16* Wk_bf  = (bf16*)take((size_t)DM * DM * 2);
  bf16* Wt1_bf = (bf16*)take((size_t)DB * DM * 2);
  bf16* Wt2_bf = (bf16*)take((size_t)DM * DB * 2);
  bf16* W1a_bf = (bf16*)take((size_t)DB * DM * 2);
  bf16* W1b_bf = (bf16*)take((size_t)DM * DB * 2);
  float* x_enc = (float*)take((size_t)BQ * DM * 4);
  float* k_f32 = (float*)take((size_t)BQ * DM * 4);
  bf16*  k_bf  = (bf16*)take((size_t)BQ * DM * 2);
  float* sq_k  = (float*)take((size_t)BQ * 4);
  float* ck_f  = (float*)take((size_t)NPAD * DM * 4);
  bf16*  ck_bf = (bf16*)take((size_t)NPAD * DM * 2);
  float* sq_c  = (float*)take((size_t)NPAD * 4);
  int*   idx   = (int*)take((size_t)BQ * CTX * 4);
  float* sims  = (float*)take((size_t)BQ * CTX * 4);
  float* x2    = (float*)take((size_t)BQ * DM * 4);

  // weight conversion + pad
  cvt_bf16<<<(DM * DI + 255) / 256, 256, 0, stream>>>(W_in, Win_bf, DM * DI);
  cvt_bf16<<<(DB * DM + 255) / 256, 256, 0, stream>>>(W0a, W0a_bf, DB * DM);
  cvt_bf16<<<(DM * DB + 255) / 256, 256, 0, stream>>>(W0b, W0b_bf, DM * DB);
  cvt_bf16<<<(DM * DM + 255) / 256, 256, 0, stream>>>(W_k, Wk_bf, DM * DM);
  cvt_bf16<<<(DB * DM + 255) / 256, 256, 0, stream>>>(W_t1, Wt1_bf, DB * DM);
  cvt_bf16<<<(DM * DB + 255) / 256, 256, 0, stream>>>(W_t2, Wt2_bf, DM * DB);
  cvt_bf16<<<(DB * DM + 255) / 256, 256, 0, stream>>>(W1a, W1a_bf, DB * DM);
  cvt_bf16<<<(DM * DB + 255) / 256, 256, 0, stream>>>(W1b, W1b_bf, DM * DB);
  pad_kernel<<<((NPAD - NTRAIN) * DM + 255) / 256, 256, 0, stream>>>(ck_bf, sq_c);

  // encode queries and candidates
  encode_kernel<<<BQ / 16, 128, 0, stream>>>(
      X, Win_bf, b_in, W0a_bf, b0a, W0b_bf, b0b, g_mix, be_mix, Wk_bf, b_k,
      x_enc, k_f32, k_bf, sq_k);
  encode_kernel<<<NTRAIN / 16, 128, 0, stream>>>(
      Xtr, Win_bf, b_in, W0a_bf, b0a, W0b_bf, b0b, g_mix, be_mix, Wk_bf, b_k,
      nullptr, ck_f, ck_bf, sq_c);

  // retrieval (async-staged candidate stream + WMMA scoring)
  topk_kernel<<<BQ / 16, 128, 0, stream>>>(k_bf, sq_k, ck_bf, sq_c, idx, sims);

  // context aggregation
  aggregate_kernel<<<BQ, 256, 0, stream>>>(
      k_f32, ck_f, ytr, idx, sims, Wt1_bf, b_t1, Wt2_bf, w_lab, b_lab, x_enc, x2);

  // predictor + head
  predictor_kernel<<<BQ / 16, 128, 0, stream>>>(
      x2, g_p, be_p, W1a_bf, b1a, W1b_bf, b1b, g_h, be_h, W_h, b_h, out);
}
